// DBwGcn_71519795413843
// MI455X (gfx1250) — matrix operand order repaired and checked by reading, not compiled, not run on hardware
//
#include <hip/hip_runtime.h>
#include <hip/hip_bf16.h>

#define NN 20000
#define NE 320000
#define NSAMP 25
#define BN_SCALE 0.9999950000374997f
#define COS_EPS 1e-8f

typedef __attribute__((ext_vector_type(16))) _Float16 v16h;
typedef __attribute__((ext_vector_type(8)))  _Float16 v8h;
typedef __attribute__((ext_vector_type(4)))  _Float16 v4h;
typedef __attribute__((ext_vector_type(8)))  float    v8f;

static __device__ __forceinline__ v8f vzero8() {
  v8f z;
#pragma unroll
  for (int i = 0; i < 8; ++i) z[i] = 0.0f;
  return z;
}

// Build a 16x16x32 f16 A-fragment from a row-major f16 LDS tile.
// Lane l (<16): row=l, K = {kt*32+0..7, kt*32+16..23}
// Lane l (>=16): row=l-16, K = {kt*32+8..15, kt*32+24..31}
static __device__ __forceinline__ v16h lds_a_frag(const _Float16* tile, int strideH,
                                                  int rowOff, int lane, int kt) {
  const _Float16* p = tile + (rowOff + (lane & 15)) * strideH + kt * 32 + ((lane >> 4) << 3);
  v8h lo = *(const v8h*)(p);
  v8h hi = *(const v8h*)(p + 16);
  v16h a;
#pragma unroll
  for (int i = 0; i < 8; ++i) { a[i] = lo[i]; a[i + 8] = hi[i]; }
  return a;
}

// Packed B layout: [term][kt][nt][lane][16 halfs] ; lane<16 -> K 0..7,16..23 of col nt*16+lane,
// lane>=16 -> K 8..15,24..31. One contiguous 32B load per lane.
static __device__ __forceinline__ v16h ld_b_frag(const _Float16* __restrict__ Bp,
                                                 int KT, int NT, int term, int kt, int nt, int lane) {
  return *(const v16h*)(Bp + ((((size_t)term * KT + kt) * NT + nt) * 32 + lane) * 16);
}

static __device__ __forceinline__ v8f wmma16(v16h a, v16h b, v8f c) {
  return __builtin_amdgcn_wmma_f32_16x16x32_f16(false, a, false, b, (short)0, c, false, false);
}

// ---------------------------------------------------------------------------
// Weight repack: f32 [out,in] (transpose=1) or [K,N] (transpose=0) -> packed f16 B
// ---------------------------------------------------------------------------
__global__ void pack_b_kernel(const float* __restrict__ src, _Float16* __restrict__ dst,
                              int KT, int NT, int srcLd, int transpose, int nTerms,
                              long termStride) {
  long total  = (long)nTerms * KT * NT * 512;
  long stride = (long)gridDim.x * blockDim.x;
  for (long idx = (long)blockIdx.x * blockDim.x + threadIdx.x; idx < total; idx += stride) {
    long per  = (long)KT * NT * 512;
    int  term = (int)(idx / per);
    long rem  = idx - (long)term * per;
    int  i    = (int)(rem & 15);
    int  lane = (int)((rem >> 4) & 31);
    int  nt   = (int)((rem >> 9) % NT);
    int  kt   = (int)(rem / ((long)NT << 9));
    int  k    = kt * 32 + ((i >> 3) << 4) + ((lane >> 4) << 3) + (i & 7);
    int  n    = nt * 16 + (lane & 15);
    float v = transpose ? src[term * termStride + (long)n * srcLd + k]
                        : src[term * termStride + (long)k * srcLd + n];
    dst[idx] = (_Float16)v;
  }
}

// ---------------------------------------------------------------------------
// Generic node GEMM: C = epi( sum_t A_t[N,F] @ B_t[F,NOUT] + bias )
// block = 128 thr (4 waves), 64 rows/block; A staged f32->f16 via LDS.
// mode: 0 = +bias, 1 = relu(+bias), 2 = relu(+bias)*BN_SCALE
// ---------------------------------------------------------------------------
template <int F, int NOUT, int NTERMS>
__global__ void __launch_bounds__(128) gemm_node_kernel(
    const float* __restrict__ A0, const float* __restrict__ A1,
    const float* __restrict__ A2, const float* __restrict__ A3, int lda,
    const _Float16* __restrict__ Bp, const float* __restrict__ bias,
    float* __restrict__ C1, int ldc1, float* __restrict__ C2, int ldc2,
    int nRows, int mode) {
  constexpr int KT = F / 32;
  constexpr int NT = NOUT / 16;
  __shared__ __align__(16) _Float16 At[64 * 40];
  int t = threadIdx.x, wave = t >> 5, lane = t & 31;
  int rowBase = blockIdx.x * 64;
  v8f acc[NT];
#pragma unroll
  for (int i = 0; i < NT; ++i) acc[i] = vzero8();

#pragma unroll
  for (int term = 0; term < NTERMS; ++term) {
    const float* A = (term == 0) ? A0 : (term == 1) ? A1 : (term == 2) ? A2 : A3;
    for (int kt = 0; kt < KT; ++kt) {
      __syncthreads();
      // cooperative 64x32 f32 -> f16 stage
#pragma unroll
      for (int i = 0; i < 4; ++i) {
        int f4 = t + 128 * i;           // 0..511 float4 slots
        int r = f4 >> 3, c4 = f4 & 7;   // 64 rows x 8 float4
        int gr = rowBase + r;
        float4 v = make_float4(0.f, 0.f, 0.f, 0.f);
        if (gr < nRows) v = *(const float4*)(A + (size_t)gr * lda + kt * 32 + c4 * 4);
        v4h h;
        h[0] = (_Float16)v.x; h[1] = (_Float16)v.y; h[2] = (_Float16)v.z; h[3] = (_Float16)v.w;
        *(v4h*)(&At[r * 40 + c4 * 4]) = h;
      }
      __syncthreads();
      v16h a = lds_a_frag(At, 40, wave * 16, lane, 0);
#pragma unroll
      for (int nt = 0; nt < NT; ++nt) {
        v16h b = ld_b_frag(Bp, KT, NT, term, kt, nt, lane);
        acc[nt] = wmma16(a, b, acc[nt]);
      }
    }
  }

  int cLane = lane & 15;
  int rOff  = (lane >> 4) * 8;
#pragma unroll
  for (int nt = 0; nt < NT; ++nt) {
    int col = nt * 16 + cLane;
    float bv = bias ? bias[col] : 0.f;
#pragma unroll
    for (int i = 0; i < 8; ++i) {
      int row = rowBase + wave * 16 + rOff + i;
      if (row < nRows) {
        float v = acc[nt][i] + bv;
        if (mode >= 1) v = fmaxf(v, 0.f);
        if (mode == 2) v *= BN_SCALE;
        C1[(size_t)row * ldc1 + col] = v;
        if (C2) C2[(size_t)row * ldc2 + col] = v;
      }
    }
  }
}

// ---------------------------------------------------------------------------
// Fused edge MLP (64->512->256->128) for both input halves + cosine -> ew[e]
// block = 128 thr (4 waves) handles 16 edges = 32 MLP rows. All layers in LDS.
// GEMM1/GEMM2 are split over K halves so h1 needs only 32x264 f16.
// ---------------------------------------------------------------------------
__global__ void __launch_bounds__(128) edge_mlp_kernel(
    const float* __restrict__ ein,
    const _Float16* __restrict__ pB1, const _Float16* __restrict__ pB2,
    const _Float16* __restrict__ pB3,
    const float* __restrict__ be1, const float* __restrict__ be2,
    const float* __restrict__ be3, float* __restrict__ ew) {
  __shared__ __align__(16) _Float16 inA[32 * 72];
  __shared__ __align__(16) _Float16 h1[32 * 264];
  __shared__ __align__(16) _Float16 h2[32 * 264];
  __shared__ __align__(16) float    out3[32 * 132];
  int t = threadIdx.x, wave = t >> 5, lane = t & 31;
  int eBase = blockIdx.x * 16;

  // stage inputs: row 2i = edge half0 (cols 0:64), row 2i+1 = half1 (cols 64:128)
#pragma unroll
  for (int i = 0; i < 4; ++i) {
    int f4 = t + 128 * i;           // 0..511 : 32 rows x 16 float4
    int r = f4 >> 4, c4 = f4 & 15;
    int e  = eBase + (r >> 1);
    int cb = (r & 1) * 64;
    float4 v = *(const float4*)(ein + (size_t)e * 128 + cb + c4 * 4);
    v4h h;
    h[0] = (_Float16)v.x; h[1] = (_Float16)v.y; h[2] = (_Float16)v.z; h[3] = (_Float16)v.w;
    *(v4h*)(&inA[r * 72 + c4 * 4]) = h;
  }
  __syncthreads();

  int rt = wave & 1;   // row tile (16 rows)
  int cg = wave >> 1;  // column group
  v16h ain0 = lds_a_frag(inA, 72, rt * 16, lane, 0);
  v16h ain1 = lds_a_frag(inA, 72, rt * 16, lane, 1);

  v8f acc2[8];
#pragma unroll
  for (int j = 0; j < 8; ++j) acc2[j] = vzero8();

  for (int half = 0; half < 2; ++half) {
    __syncthreads();  // protect h1 overwrite vs previous half's reads
    // ---- GEMM1: this half's 256 output cols -> h1 (relu * BN, f16) ----
#pragma unroll
    for (int j = 0; j < 8; ++j) {
      int ctl = cg * 8 + j;            // local col tile 0..15
      int ct  = half * 16 + ctl;       // global col tile 0..31
      v8f a1 = vzero8();
      a1 = wmma16(ain0, ld_b_frag(pB1, 2, 32, 0, 0, ct, lane), a1);
      a1 = wmma16(ain1, ld_b_frag(pB1, 2, 32, 0, 1, ct, lane), a1);
      int cc = ctl * 16 + (lane & 15);
      float bv = be1[half * 256 + cc];
      int rb = rt * 16 + ((lane >> 4) << 3);
#pragma unroll
      for (int i = 0; i < 8; ++i)
        h1[(rb + i) * 264 + cc] = (_Float16)(fmaxf(a1[i] + bv, 0.f) * BN_SCALE);
    }
    __syncthreads();
    // ---- GEMM2 partial accumulate over this K half ----
    for (int ktl = 0; ktl < 8; ++ktl) {
      v16h a = lds_a_frag(h1, 264, rt * 16, lane, ktl);
#pragma unroll
      for (int j = 0; j < 8; ++j) {
        v16h b = ld_b_frag(pB2, 16, 16, 0, half * 8 + ktl, cg * 8 + j, lane);
        acc2[j] = wmma16(a, b, acc2[j]);
      }
    }
  }

  // ---- GEMM2 epilogue -> h2 (relu * BN, f16) ----
#pragma unroll
  for (int j = 0; j < 8; ++j) {
    int cc = (cg * 8 + j) * 16 + (lane & 15);
    float bv = be2[cc];
    int rb = rt * 16 + ((lane >> 4) << 3);
#pragma unroll
    for (int i = 0; i < 8; ++i)
      h2[(rb + i) * 264 + cc] = (_Float16)(fmaxf(acc2[j][i] + bv, 0.f) * BN_SCALE);
  }
  __syncthreads();

  // ---- GEMM3: 256 -> 128, f32 out ----
  v8f acc3[4];
#pragma unroll
  for (int j = 0; j < 4; ++j) acc3[j] = vzero8();
  for (int kt = 0; kt < 8; ++kt) {
    v16h a = lds_a_frag(h2, 264, rt * 16, lane, kt);
#pragma unroll
    for (int j = 0; j < 4; ++j) {
      v16h b = ld_b_frag(pB3, 8, 8, 0, kt, cg * 4 + j, lane);
      acc3[j] = wmma16(a, b, acc3[j]);
    }
  }
#pragma unroll
  for (int j = 0; j < 4; ++j) {
    int cc = (cg * 4 + j) * 16 + (lane & 15);
    float bv = be3[cc];
    int rb = rt * 16 + ((lane >> 4) << 3);
#pragma unroll
    for (int i = 0; i < 8; ++i) out3[(rb + i) * 132 + cc] = acc3[j][i] + bv;
  }
  __syncthreads();

  // ---- cosine head: 8 lanes per edge ----
  int el = t >> 3, part = t & 7;
  const float* ra = &out3[(2 * el) * 132];
  const float* rb = &out3[(2 * el + 1) * 132];
  float d = 0.f, na = 0.f, nb = 0.f;
#pragma unroll
  for (int i = 0; i < 16; ++i) {
    float x = ra[part * 16 + i], y = rb[part * 16 + i];
    d += x * y; na += x * x; nb += y * y;
  }
#pragma unroll
  for (int m = 1; m < 8; m <<= 1) {
    d  += __shfl_xor(d,  m, 32);
    na += __shfl_xor(na, m, 32);
    nb += __shfl_xor(nb, m, 32);
  }
  if (part == 0) {
    float n1 = fmaxf(sqrtf(na), COS_EPS);
    float n2 = fmaxf(sqrtf(nb), COS_EPS);
    ew[eBase + el] = (d / (n1 * n2) + 1.f) * 0.5f;
  }
}

// ---------------------------------------------------------------------------
// Graph / elementwise helper kernels
// ---------------------------------------------------------------------------
__global__ void fill_zero_kernel(float* __restrict__ p, long n) {
  long s = (long)gridDim.x * blockDim.x;
  for (long i = (long)blockIdx.x * blockDim.x + threadIdx.x; i < n; i += s) p[i] = 0.f;
}

__global__ void copy_kernel(float* __restrict__ dst, const float* __restrict__ src, long n) {
  long s = (long)gridDim.x * blockDim.x;
  for (long i = (long)blockIdx.x * blockDim.x + threadIdx.x; i < n; i += s) dst[i] = src[i];
}

__global__ void axpy_kernel(float* __restrict__ y, const float* __restrict__ x, float a, long n) {
  long s = (long)gridDim.x * blockDim.x;
  for (long i = (long)blockIdx.x * blockDim.x + threadIdx.x; i < n; i += s) y[i] += a * x[i];
}

__global__ void deg_kernel(const int* __restrict__ row, const int* __restrict__ col,
                           const float* __restrict__ ew, float* __restrict__ deg, int E) {
  int s = gridDim.x * blockDim.x;
  for (int e = blockIdx.x * blockDim.x + threadIdx.x; e < E; e += s) {
    int r = row[e], c = col[e];
    float w = (r == c) ? 0.f : ew[e];
    atomicAdd(&deg[r], w);
  }
}

__global__ void dinv_kernel(const float* __restrict__ deg, float* __restrict__ dinv, int n) {
  int s = gridDim.x * blockDim.x;
  for (int i = blockIdx.x * blockDim.x + threadIdx.x; i < n; i += s) {
    float d = deg[i];
    dinv[i] = (d > 0.f) ? rsqrtf(d) : 0.f;
  }
}

__global__ void lapw_kernel(const int* __restrict__ row, const int* __restrict__ col,
                            const float* __restrict__ ew, const float* __restrict__ dinv,
                            float* __restrict__ lapw, int E) {
  int s = gridDim.x * blockDim.x;
  for (int e = blockIdx.x * blockDim.x + threadIdx.x; e < E; e += s) {
    int r = row[e], c = col[e];
    float w = (r == c) ? 0.f : ew[e];
    lapw[e] = -dinv[r] * w * dinv[c];
  }
}

// y[col[e], :] += scale * w[e] * x[row[e], :], vectorized float4 per thread
template <int F>
__global__ void scatter_kernel(const int* __restrict__ row, const int* __restrict__ col,
                               const float* __restrict__ w, const float* __restrict__ x,
                               float* __restrict__ y, float scale, int E) {
  constexpr int CH = F / 4;
  long total  = (long)E * CH;
  long stride = (long)gridDim.x * blockDim.x;
  for (long i = (long)blockIdx.x * blockDim.x + threadIdx.x; i < total; i += stride) {
    int e = (int)(i / CH);
    int c = (int)(i - (long)e * CH);
    float wv = w[e] * scale;
    int r = row[e], cc = col[e];
    float4 xv = *(const float4*)(x + (size_t)r * F + c * 4);
    float* yp = y + (size_t)cc * F + c * 4;
    atomicAdd(yp + 0, wv * xv.x);
    atomicAdd(yp + 1, wv * xv.y);
    atomicAdd(yp + 2, wv * xv.z);
    atomicAdd(yp + 3, wv * xv.w);
  }
}

__global__ void logit1_kernel(const float* __restrict__ m, const float* __restrict__ Wm2,
                              const float* __restrict__ bm2, float* __restrict__ out, int n) {
  int s = gridDim.x * blockDim.x;
  for (int node = blockIdx.x * blockDim.x + threadIdx.x; node < n; node += s) {
    float a0 = bm2[0], a1 = bm2[1];
    const float* mr = m + (size_t)node * 256;
#pragma unroll 4
    for (int j = 0; j < 64; ++j) {
      float4 mv = *(const float4*)(mr + j * 4);
      float4 w0 = *(const float4*)(Wm2 + j * 4);
      float4 w1 = *(const float4*)(Wm2 + 256 + j * 4);
      a0 += mv.x * w0.x + mv.y * w0.y + mv.z * w0.z + mv.w * w0.w;
      a1 += mv.x * w1.x + mv.y * w1.y + mv.z * w1.z + mv.w * w1.w;
    }
    out[(size_t)node * 2 + 0] = a0;
    out[(size_t)node * 2 + 1] = a1;
  }
}

__global__ void logit2_kernel(const float* __restrict__ states, const float* __restrict__ Whl,
                              const float* __restrict__ bhl, float* __restrict__ out, int n) {
  int stride = gridDim.x * blockDim.x;
  for (int node = blockIdx.x * blockDim.x + threadIdx.x; node < n; node += stride) {
#pragma unroll
    for (int c = 0; c < 2; ++c) {
      float acc = bhl[c];
      for (int s = 0; s < NSAMP; ++s) {
        const float* sp = states + ((size_t)s * NN + node) * 16;
        const float* wp = Whl + c * 400 + s * 16;
#pragma unroll
        for (int j = 0; j < 4; ++j) {
          float4 sv = *(const float4*)(sp + j * 4);
          float4 wv = *(const float4*)(wp + j * 4);
          acc += sv.x * wv.x + sv.y * wv.y + sv.z * wv.z + sv.w * wv.w;
        }
      }
      out[(size_t)node * 2 + c] = 1.f / (1.f + expf(-acc));
    }
  }
}

__global__ void final_kernel(const float* __restrict__ l1, const float* __restrict__ l2,
                             float* __restrict__ out, int n) {
  int s = gridDim.x * blockDim.x;
  for (int node = blockIdx.x * blockDim.x + threadIdx.x; node < n; node += s) {
    float a = l1[2 * node], b = l1[2 * node + 1];
    float mx = fmaxf(a, b);
    float ea = expf(a - mx), eb = expf(b - mx);
    float inv = 1.f / (ea + eb);
    float p1a = ea * inv, p1b = eb * inv;
    a = l2[2 * node]; b = l2[2 * node + 1];
    mx = fmaxf(a, b);
    ea = expf(a - mx); eb = expf(b - mx);
    inv = 1.f / (ea + eb);
    float p2a = ea * inv, p2b = eb * inv;
    out[2 * node]                 = p1a;
    out[2 * node + 1]             = p1b;
    out[2 * (size_t)n + 2 * node]     = p2a;
    out[2 * (size_t)n + 2 * node + 1] = p2b;
    out[4 * (size_t)n + 2 * node]     = fmaxf(p1a, p2a);
    out[4 * (size_t)n + 2 * node + 1] = fmaxf(p1b, p2b);
  }
}

static inline unsigned cdiv(long a, long b) { return (unsigned)((a + b - 1) / b); }

extern "C" void kernel_launch(void* const* d_in, const int* in_sizes, int n_in,
                              void* d_out, int out_size, void* d_ws, size_t ws_size,
                              hipStream_t stream) {
  (void)in_sizes; (void)n_in; (void)out_size; (void)ws_size;
  const float* features = (const float*)d_in[0];
  const int*   eidx     = (const int*)d_in[1];
  const float* ein      = (const float*)d_in[2];
  const float* We1 = (const float*)d_in[3];  const float* be1 = (const float*)d_in[4];
  const float* We2 = (const float*)d_in[5];  const float* be2 = (const float*)d_in[6];
  const float* We3 = (const float*)d_in[7];  const float* be3 = (const float*)d_in[8];
  const float* cw0 = (const float*)d_in[9];
  const float* cw1 = (const float*)d_in[10];
  const float* cw2 = (const float*)d_in[11];
  const float* cw3 = (const float*)d_in[12];
  const float* Wm1 = (const float*)d_in[13]; const float* bm1 = (const float*)d_in[14];
  const float* Wm2 = (const float*)d_in[15]; const float* bm2 = (const float*)d_in[16];
  const float* Wl  = (const float*)d_in[17]; const float* bl  = (const float*)d_in[18];
  const float* Whl = (const float*)d_in[19]; const float* bhl = (const float*)d_in[20];
  const float* noise = (const float*)d_in[21];
  const int* erow = eidx;
  const int* ecol = eidx + NE;

  // ---- bump allocator over d_ws ----
  char* ws = (char*)d_ws;
  size_t off = 0;
  auto alloc = [&](size_t bytes) -> void* {
    off = (off + 255) & ~(size_t)255;
    void* p = ws + off;
    off += bytes;
    return p;
  };
  _Float16* pB1  = (_Float16*)alloc((size_t)64 * 512 * 2);
  _Float16* pB2  = (_Float16*)alloc((size_t)512 * 256 * 2);
  _Float16* pB3  = (_Float16*)alloc((size_t)256 * 128 * 2);
  _Float16* pC0  = (_Float16*)alloc((size_t)4 * 128 * 64 * 2);
  _Float16* pC1  = (_Float16*)alloc((size_t)4 * 64 * 64 * 2);
  _Float16* pC2  = (_Float16*)alloc((size_t)4 * 64 * 64 * 2);
  _Float16* pC3  = (_Float16*)alloc((size_t)4 * 64 * 64 * 2);
  _Float16* pWm1 = (_Float16*)alloc((size_t)256 * 256 * 2);
  _Float16* pWl  = (_Float16*)alloc((size_t)128 * 16 * 2);
  float* ew   = (float*)alloc((size_t)NE * 4);
  float* deg  = (float*)alloc((size_t)NN * 4);
  float* dinv = (float*)alloc((size_t)NN * 4);
  float* lapw = (float*)alloc((size_t)NE * 4);
  float* jk   = (float*)alloc((size_t)NN * 256 * 4);
  float* lg1  = (float*)alloc((size_t)NN * 2 * 4);
  float* lg2  = (float*)alloc((size_t)NN * 2 * 4);
  float* h0   = (float*)alloc((size_t)NN * 16 * 4);
  // big region time-multiplexed: Cheb tx0..tx3 -> classifier m -> Bayesian states
  char* big = (char*)alloc((size_t)4 * NN * 128 * 4);
  float* tx0 = (float*)big;
  float* tx1 = tx0 + (size_t)NN * 128;
  float* tx2 = tx1 + (size_t)NN * 128;
  float* tx3 = tx2 + (size_t)NN * 128;
  float* mbuf   = (float*)big;
  float* states = (float*)big;

  // ---- pack all GEMM weights into CDNA5 WMMA B-fragment layout (f16) ----
  pack_b_kernel<<<128, 256, 0, stream>>>(We1, pB1, 2, 32, 64, 1, 1, 0);
  pack_b_kernel<<<512, 256, 0, stream>>>(We2, pB2, 16, 16, 512, 1, 1, 0);
  pack_b_kernel<<<128, 256, 0, stream>>>(We3, pB3, 8, 8, 256, 1, 1, 0);
  pack_b_kernel<<<128, 256, 0, stream>>>(cw0, pC0, 4, 4, 64, 0, 4, 128 * 64);
  pack_b_kernel<<<64, 256, 0, stream>>>(cw1, pC1, 2, 4, 64, 0, 4, 64 * 64);
  pack_b_kernel<<<64, 256, 0, stream>>>(cw2, pC2, 2, 4, 64, 0, 4, 64 * 64);
  pack_b_kernel<<<64, 256, 0, stream>>>(cw3, pC3, 2, 4, 64, 0, 4, 64 * 64);
  pack_b_kernel<<<256, 256, 0, stream>>>(Wm1, pWm1, 8, 16, 256, 1, 1, 0);
  pack_b_kernel<<<8, 256, 0, stream>>>(Wl, pWl, 4, 1, 128, 1, 1, 0);

  // ---- fused edge MLP + cosine -> edge weights ----
  edge_mlp_kernel<<<NE / 16, 128, 0, stream>>>(ein, pB1, pB2, pB3, be1, be2, be3, ew);

  // ---- scaled Laplacian off-diagonal weights ----
  fill_zero_kernel<<<cdiv(NN, 256), 256, 0, stream>>>(deg, NN);
  deg_kernel<<<cdiv(NE, 256), 256, 0, stream>>>(erow, ecol, ew, deg, NE);
  dinv_kernel<<<cdiv(NN, 256), 256, 0, stream>>>(deg, dinv, NN);
  lapw_kernel<<<cdiv(NE, 256), 256, 0, stream>>>(erow, ecol, ew, dinv, lapw, NE);

  // ---- ChebConv stack (K=4) with jumping-knowledge concat ----
  const float* xcur = features;
  for (int L = 0; L < 4; ++L) {
    const int F = (L == 0) ? 128 : 64;
    const long nf = (long)NN * F;
    // tx1 = L_hat @ x
    fill_zero_kernel<<<cdiv(nf, 256), 256, 0, stream>>>(tx1, nf);
    if (F == 128) scatter_kernel<128><<<cdiv((long)NE * 32, 256), 256, 0, stream>>>(erow, ecol, lapw, xcur, tx1, 1.f, NE);
    else          scatter_kernel<64><<<cdiv((long)NE * 16, 256), 256, 0, stream>>>(erow, ecol, lapw, xcur, tx1, 1.f, NE);
    // tx2 = 2*L_hat@tx1 - x
    fill_zero_kernel<<<cdiv(nf, 256), 256, 0, stream>>>(tx2, nf);
    if (F == 128) scatter_kernel<128><<<cdiv((long)NE * 32, 256), 256, 0, stream>>>(erow, ecol, lapw, tx1, tx2, 2.f, NE);
    else          scatter_kernel<64><<<cdiv((long)NE * 16, 256), 256, 0, stream>>>(erow, ecol, lapw, tx1, tx2, 2.f, NE);
    axpy_kernel<<<cdiv(nf, 256), 256, 0, stream>>>(tx2, xcur, -1.f, nf);
    // tx3 = 2*L_hat@tx2 - tx1
    fill_zero_kernel<<<cdiv(nf, 256), 256, 0, stream>>>(tx3, nf);
    if (F == 128) scatter_kernel<128><<<cdiv((long)NE * 32, 256), 256, 0, stream>>>(erow, ecol, lapw, tx2, tx3, 2.f, NE);
    else          scatter_kernel<64><<<cdiv((long)NE * 16, 256), 256, 0, stream>>>(erow, ecol, lapw, tx2, tx3, 2.f, NE);
    axpy_kernel<<<cdiv(nf, 256), 256, 0, stream>>>(tx3, tx1, -1.f, nf);
    // h = relu( sum_k Txk @ Wk )  -> next input (tx0) and jk[:, 64L:64L+64]
    if (L == 0)
      gemm_node_kernel<128, 64, 4><<<cdiv(NN, 64), 128, 0, stream>>>(
          xcur, tx1, tx2, tx3, 128, pC0, nullptr, tx0, 64, jk + 64 * L, 256, NN, 1);
    else {
      const _Float16* pc = (L == 1) ? pC1 : (L == 2) ? pC2 : pC3;
      gemm_node_kernel<64, 64, 4><<<cdiv(NN, 64), 128, 0, stream>>>(
          xcur, tx1, tx2, tx3, 64, pc, nullptr, tx0, 64, jk + 64 * L, 256, NN, 1);
    }
    xcur = tx0;
  }

  // ---- classifier: m = relu(jk @ Wm1^T + bm1)*BN ; logit1 = m @ Wm2^T + bm2 ----
  gemm_node_kernel<256, 256, 1><<<cdiv(NN, 64), 128, 0, stream>>>(
      jk, nullptr, nullptr, nullptr, 256, pWm1, bm1, mbuf, 256, nullptr, 0, NN, 2);
  logit1_kernel<<<cdiv(NN, 256), 256, 0, stream>>>(mbuf, Wm2, bm2, lg1, NN);

  // ---- Bayesian branch: h0 = X @ Wl^T + bl; 25 chained noise+scatter steps ----
  gemm_node_kernel<128, 16, 1><<<cdiv(NN, 64), 128, 0, stream>>>(
      features, nullptr, nullptr, nullptr, 128, pWl, bl, h0, 16, nullptr, 0, NN, 0);
  const float* hc = h0;
  for (int s = 0; s < NSAMP; ++s) {
    float* st = states + (size_t)s * NN * 16;
    copy_kernel<<<cdiv((long)NN * 16, 256), 256, 0, stream>>>(st, noise + (size_t)s * NN * 16, (long)NN * 16);
    scatter_kernel<16><<<cdiv((long)NE * 4, 256), 256, 0, stream>>>(erow, ecol, ew, hc, st, 1.f, NE);
    hc = st;
  }
  logit2_kernel<<<cdiv(NN, 256), 256, 0, stream>>>(states, Whl, bhl, lg2, NN);

  // ---- softmaxes + elementwise max into d_out (p1 | p2 | max) ----
  final_kernel<<<cdiv(NN, 256), 256, 0, stream>>>(lg1, lg2, (float*)d_out, NN);
}
